// DVT_69475390980615
// MI455X (gfx1250) — matrix operand order repaired and checked by reading, not compiled, not run on hardware
//
#include <hip/hip_runtime.h>
#include <hip/hip_bf16.h>

typedef __bf16 bf16;
typedef __attribute__((ext_vector_type(16))) __bf16 v16bf;
typedef __attribute__((ext_vector_type(8)))  __bf16 v8bf;
typedef __attribute__((ext_vector_type(8)))  float  v8f;
typedef __attribute__((ext_vector_type(4)))  unsigned int u32x4;
typedef __attribute__((ext_vector_type(8)))  int i32x8;
typedef __attribute__((ext_vector_type(4)))  int i32x4;

#define BM 128
#define BN 64
#define LDP2 72          // padded row (144B = multiple of 16B) for BK=64 tiles
#define LDP1 40          // padded row for the single BK=32 kernel (dots2)

#if defined(__gfx1250__) && __has_builtin(__builtin_amdgcn_tensor_load_to_lds) && \
    __has_builtin(__builtin_amdgcn_s_wait_tensorcnt)
#define USE_TDM 1
#endif

__device__ __forceinline__ v8f wmma_bf16(v16bf a, v16bf b, v8f c) {
  return __builtin_amdgcn_wmma_f32_16x16x32_bf16(false, a, false, b, (short)0, c, false, false);
}

__device__ __forceinline__ v16bf frag2(const bf16* base, int off0, int off1) {
  v8bf lo = *(const v8bf*)(base + off0);
  v8bf hi = *(const v8bf*)(base + off1);
  return __builtin_shufflevector(lo, hi, 0,1,2,3,4,5,6,7,8,9,10,11,12,13,14,15);
}

// A fragment 16x32 bf16 (ISA 7.12.2): lane<16 -> row=lane,    K in {0..7, 16..23}
//                                     lane>=16 -> row=lane-16, K in {8..15, 24..31}
template <int LD>
__device__ __forceinline__ v16bf lds_frag_a(const bf16 (*As)[LD], int mBase, int kh, int lane) {
  const bf16* rp = As[mBase + (lane & 15)] + kh;
  int kb = (lane & 16) ? 8 : 0;
  return frag2(rp, kb, kb + 16);
}
// B fragment 32x16 bf16, LDS stored transposed Bt[n][k] -> same K order as A fragment
template <int LD>
__device__ __forceinline__ v16bf lds_frag_b(const bf16 (*Bt)[LD], int nBase, int kh, int lane) {
  const bf16* rp = Bt[nBase + (lane & 15)] + kh;
  int kb = (lane & 16) ? 8 : 0;
  return frag2(rp, kb, kb + 16);
}

#define WMMA_STEP(AS, BT, LD)                                        \
  _Pragma("unroll")                                                  \
  for (int kh = 0; kh < 64; kh += 32) {                              \
    v16bf a0 = lds_frag_a<LD>(AS, wm,      kh, lane);                \
    v16bf a1 = lds_frag_a<LD>(AS, wm + 16, kh, lane);                \
    v16bf b0 = lds_frag_b<LD>(BT, wn,      kh, lane);                \
    v16bf b1 = lds_frag_b<LD>(BT, wn + 16, kh, lane);                \
    acc[0][0] = wmma_bf16(a0, b0, acc[0][0]);                        \
    acc[0][1] = wmma_bf16(a0, b1, acc[0][1]);                        \
    acc[1][0] = wmma_bf16(a1, b0, acc[1][0]);                        \
    acc[1][1] = wmma_bf16(a1, b1, acc[1][1]);                        \
  }

#ifdef USE_TDM
// Issue one TDM 2-D tile load: `rows` rows of 64 bf16 elements, global row
// stride `strideElems` (bf16), into LDS offset lds_off with a 4-DWORD pad
// every 32 DWORDs (=> LDS row pitch 72 bf16, matching [rows][LDP2]).
__device__ __forceinline__ void tdm_load_2d(const void* gptr, unsigned lds_off,
                                            unsigned rows, unsigned strideElems) {
  unsigned long long ga = (unsigned long long)(uintptr_t)gptr;
  u32x4 g0;
  g0[0] = 1u;                                   // count=1, user descriptor
  g0[1] = lds_off;                              // lds_addr (bytes)
  g0[2] = (unsigned)ga;                         // global_addr[31:0]
  g0[3] = (unsigned)(ga >> 32) | (2u << 30);    // global_addr[56:32] | type=2
  i32x8 g1;
  g1[0] = (int)((1u << 16)      // data_size = 2 bytes
              | (1u << 20)      // pad_enable
              | (4u << 22)      // pad_interval: 32 DWORDs
              | (3u << 25));    // pad_amount:   4 DWORDs
  g1[1] = (int)(64u << 16);     // tensor_dim0 = 64
  g1[2] = (int)(rows << 16);    // tensor_dim1 = rows
  g1[3] = (int)(64u << 16);     // tile_dim0 = 64
  g1[4] = (int)rows;            // tile_dim1 = rows (tile_dim2 = 0)
  g1[5] = (int)strideElems;     // tensor_dim0_stride
  g1[6] = 0; g1[7] = 0;
  i32x4 gz = {0, 0, 0, 0};
#if __has_include(<hip/amd_detail/amd_gfx1250_TDM.h>)
  i32x8 gz8 = {0, 0, 0, 0, 0, 0, 0, 0};
  __builtin_amdgcn_tensor_load_to_lds(g0, g1, gz, gz, gz8, 0);
#else
  __builtin_amdgcn_tensor_load_to_lds(g0, g1, gz, gz, 0);
#endif
}
#endif

// ---------------------------------------------------------------------------
// Kernel 1: fused QKV projection.  P[b,m,n] = BN( Wcat[m,:] . x[b,:,n] )
// rows [0,256)=Wq [256,512)=Wk [512,1024)=Wv.  K=256.
// ---------------------------------------------------------------------------
__global__ __launch_bounds__(256)
void k_proj(const float* __restrict__ x,
            const float* __restrict__ Wq, const float* __restrict__ gq, const float* __restrict__ bq,
            const float* __restrict__ Wk, const float* __restrict__ gk, const float* __restrict__ bk,
            const float* __restrict__ Wv, const float* __restrict__ gv, const float* __restrict__ bv,
            float* __restrict__ pf, bf16* __restrict__ pb)
{
  const int b = blockIdx.z;
  const int m0 = blockIdx.x * BM, n0 = blockIdx.y * BN;
  const int tid = threadIdx.x, lane = tid & 31, wid = tid >> 5;
  const int wm = (wid >> 1) * 32, wn = (wid & 1) * 32;
  __shared__ __align__(16) bf16 As[BM][LDP2];
  __shared__ __align__(16) bf16 Bt[BN][LDP2];
  v8f acc[2][2] = {};

  for (int k0 = 0; k0 < 256; k0 += 64) {
    for (int u = tid; u < BM * 64; u += 256) {
      int m = u >> 6, k = u & 63, gm = m0 + m, gk = k0 + k;
      float w;
      if (gm < 256)      w = Wq[gm * 256 + gk];
      else if (gm < 512) w = Wk[(gm - 256) * 256 + gk];
      else               w = Wv[(gm - 512) * 256 + gk];
      As[m][k] = (bf16)w;
    }
    for (int u = tid; u < BN * 64; u += 256) {
      int n = u & 63, k = u >> 6;
      Bt[n][k] = (bf16)x[((size_t)b * 256 + (k0 + k)) * 1024 + n0 + n];
    }
    if (k0 + 64 < 256)
      __builtin_prefetch(&x[((size_t)b * 256 + (k0 + 64)) * 1024 + n0 + (tid & 63)], 0, 0);
    __syncthreads();
    WMMA_STEP(As, Bt, LDP2)
    __syncthreads();
  }

  const int col = lane & 15, rb = (lane & 16) ? 8 : 0;
  #pragma unroll
  for (int i = 0; i < 2; i++)
    #pragma unroll
    for (int j = 0; j < 2; j++) {
      int mB = m0 + wm + i * 16 + rb;
      int n  = n0 + wn + j * 16 + col;
      #pragma unroll
      for (int r = 0; r < 8; r++) {
        int m = mB + r;
        float g, be;
        if (m < 256)      { g = gq[m];       be = bq[m]; }
        else if (m < 512) { g = gk[m - 256]; be = bk[m - 256]; }
        else              { g = gv[m - 512]; be = bv[m - 512]; }
        float v = acc[i][j][r] * g + be;
        size_t o = ((size_t)b * 1024 + m) * 1024 + n;
        pf[o] = v;
        pb[o] = (bf16)v;
      }
    }
}

// ---------------------------------------------------------------------------
// Kernel 2: dots1 = BN(Wmk @ q1) + rel-pos bias -> logits[b,h,i,j].
// M=8192 (m = h*1024 + j), N=1024 tokens (col = i), K=256.
// ---------------------------------------------------------------------------
__global__ __launch_bounds__(256)
void k_dots1(const float* __restrict__ Wmk, const float* __restrict__ gmk,
             const float* __restrict__ bmk, const float* __restrict__ pos_emb,
             const float* __restrict__ q1, float* __restrict__ logits)
{
  const int b = blockIdx.z;
  const int m0 = blockIdx.x * BM, n0 = blockIdx.y * BN;
  const int tid = threadIdx.x, lane = tid & 31, wid = tid >> 5;
  const int wm = (wid >> 1) * 32, wn = (wid & 1) * 32;
  __shared__ __align__(16) bf16 As[BM][LDP2];
  __shared__ __align__(16) bf16 Bt[BN][LDP2];
  v8f acc[2][2] = {};

  for (int k0 = 0; k0 < 256; k0 += 64) {
    for (int u = tid; u < BM * 64; u += 256) {
      int m = u >> 6, k = u & 63;
      As[m][k] = (bf16)Wmk[(size_t)(m0 + m) * 256 + k0 + k];
    }
    for (int u = tid; u < BN * 64; u += 256) {
      int n = u & 63, k = u >> 6;
      Bt[n][k] = (bf16)q1[((size_t)b * 1024 + (k0 + k)) * 1024 + n0 + n];
    }
    if (k0 + 64 < 256)
      __builtin_prefetch(&Wmk[(size_t)(m0 + (tid >> 1)) * 256 + k0 + 64], 0, 0);
    __syncthreads();
    WMMA_STEP(As, Bt, LDP2)
    __syncthreads();
  }

  const int col = lane & 15, rb = (lane & 16) ? 8 : 0;
  #pragma unroll
  for (int i = 0; i < 2; i++)
    #pragma unroll
    for (int j = 0; j < 2; j++) {
      int mB = m0 + wm + i * 16 + rb;
      int n  = n0 + wn + j * 16 + col;     // token index i
      int ix = n >> 5, iy = n & 31;
      #pragma unroll
      for (int r = 0; r < 8; r++) {
        int m = mB + r;
        int h = m >> 10, jj = m & 1023;    // head, key index j
        int jx = jj >> 5, jy = jj & 31;
        int rel = (ix > jx ? ix - jx : jx - ix) * 32 + (iy > jy ? iy - jy : jy - iy);
        float v = acc[i][j][r] * gmk[m] + bmk[m]
                + pos_emb[rel * 8 + h] * 5.656854249492381f;  // /SCALE
        logits[(((size_t)b * 8 + h) * 1024 + n) * 1024 + jj] = v;
      }
    }
}

// ---------------------------------------------------------------------------
// Kernel 3: row softmax over 1024 f32; writes bf16 attn IN PLACE
// (bf16 row occupies first 2048 bytes of each 4096-byte f32 row).
// ---------------------------------------------------------------------------
__global__ __launch_bounds__(256)
void k_softmax(float* __restrict__ logits)
{
  float* p = logits + (size_t)blockIdx.x * 1024;
  bf16*  o = (bf16*)p;
  const int t = threadIdx.x;
  float v0 = p[t], v1 = p[t + 256], v2 = p[t + 512], v3 = p[t + 768];
  float mx = fmaxf(fmaxf(v0, v1), fmaxf(v2, v3));
  __shared__ float red[8];
  #pragma unroll
  for (int off = 16; off; off >>= 1) mx = fmaxf(mx, __shfl_xor(mx, off));
  if ((t & 31) == 0) red[t >> 5] = mx;
  __syncthreads();
  if (t == 0) {
    float m2 = red[0];
    #pragma unroll
    for (int i = 1; i < 8; i++) m2 = fmaxf(m2, red[i]);
    red[0] = m2;
  }
  __syncthreads();
  mx = red[0];
  float e0 = __expf(v0 - mx), e1 = __expf(v1 - mx);
  float e2 = __expf(v2 - mx), e3 = __expf(v3 - mx);
  float s = (e0 + e1) + (e2 + e3);
  #pragma unroll
  for (int off = 16; off; off >>= 1) s += __shfl_xor(s, off);
  __syncthreads();
  if ((t & 31) == 0) red[t >> 5] = s;
  __syncthreads();
  if (t == 0) {
    float s2 = red[0];
    #pragma unroll
    for (int i = 1; i < 8; i++) s2 += red[i];
    red[0] = s2;
  }
  __syncthreads();
  float inv = 1.0f / red[0];
  o[t]       = (bf16)(e0 * inv);
  o[t + 256] = (bf16)(e1 * inv);
  o[t + 512] = (bf16)(e2 * inv);
  o[t + 768] = (bf16)(e3 * inv);
}

// ---------------------------------------------------------------------------
// Kernel 4: out = GeLU(attn @ V) per (b,h). M=1024 tok, N=64 dv, K=1024.
// Both operands are pure bf16 2-D tiles -> double-buffered Tensor Data Mover
// pipeline: wave 0 issues the next tile's two descriptor loads, then
// s_wait_tensorcnt 2 (TDM ops of one wave complete in order, so <=2
// outstanding means the previous pair has landed), barrier, compute while
// the next tile streams in.  Dynamic LDS only: D#.lds_addr = ping-pong offs.
// ---------------------------------------------------------------------------
__global__ __launch_bounds__(256)
void k_av(const float* __restrict__ logits, const bf16* __restrict__ pb,
          bf16* __restrict__ gbuf, int chOff)
{
  extern __shared__ __attribute__((aligned(16))) char smem[];
  const int bh = blockIdx.z, b = bh >> 3, h = bh & 7;
  const int m0 = blockIdx.x * BM;
  const int tid = threadIdx.x, lane = tid & 31, wid = tid >> 5;
  const int wm = (wid >> 1) * 32, wn = (wid & 1) * 32;
  const bf16* attn = (const bf16*)(logits + (size_t)bh * 1024 * 1024); // row stride 2048
  const bf16* vmat = pb + ((size_t)b * 1024 + 512 + h * 64) * 1024;    // [d][token]
  v8f acc[2][2] = {};
  const unsigned aBytes   = BM * LDP2 * (unsigned)sizeof(bf16);        // 18432
  const unsigned bufBytes = (BM + BN) * LDP2 * (unsigned)sizeof(bf16); // 27648

#ifdef USE_TDM
  if (wid == 0) {                                   // prologue: tile 0 -> buf 0
    tdm_load_2d(attn + (size_t)m0 * 2048, 0u, 128u, 2048u);
    tdm_load_2d(vmat, aBytes, 64u, 1024u);
  }
  for (int it = 0; it < 16; ++it) {
    const int k0 = it * 64;
    const unsigned cur = (it & 1) ? bufBytes : 0u;
    if (it + 1 < 16) {
      const unsigned nxt = ((it + 1) & 1) ? bufBytes : 0u;
      if (wid == 0) {
        tdm_load_2d(attn + (size_t)m0 * 2048 + k0 + 64, nxt, 128u, 2048u);
        tdm_load_2d(vmat + k0 + 64, nxt + aBytes, 64u, 1024u);
        __builtin_amdgcn_s_wait_tensorcnt(2);       // previous pair complete
      }
    } else if (wid == 0) {
      __builtin_amdgcn_s_wait_tensorcnt(0);         // drain for final tile
    }
    __syncthreads();
    const bf16 (*As)[LDP2] = (const bf16(*)[LDP2])(smem + cur);
    const bf16 (*Bt)[LDP2] = (const bf16(*)[LDP2])(smem + cur + aBytes);
    WMMA_STEP(As, Bt, LDP2)
    __syncthreads();
  }
#else
  bf16 (*As)[LDP2] = (bf16(*)[LDP2])smem;
  bf16 (*Bt)[LDP2] = (bf16(*)[LDP2])(smem + aBytes);
  for (int k0 = 0; k0 < 1024; k0 += 64) {
    for (int u = tid; u < BM * 64; u += 256) {
      int k = u & 63, m = u >> 6;
      As[m][k] = attn[(size_t)(m0 + m) * 2048 + k0 + k];
    }
    for (int u = tid; u < BN * 64; u += 256) {
      int k = u & 63, n = u >> 6;
      Bt[n][k] = vmat[(size_t)n * 1024 + k0 + k];
    }
    __syncthreads();
    WMMA_STEP(As, Bt, LDP2)
    __syncthreads();
  }
#endif
  (void)bufBytes;

  const int col = lane & 15, rb = (lane & 16) ? 8 : 0;
  #pragma unroll
  for (int i = 0; i < 2; i++)
    #pragma unroll
    for (int j = 0; j < 2; j++) {
      int tok0 = m0 + wm + i * 16 + rb;
      int d    = wn + j * 16 + col;
      int ch   = chOff + h * 64 + d;
      #pragma unroll
      for (int r = 0; r < 8; r++) {
        float v = acc[i][j][r];
        float g = 0.5f * v * (1.0f + erff(v * 0.7071067811865476f)); // exact GeLU
        gbuf[((size_t)b * 1024 + ch) * 1024 + tok0 + r] = (bf16)g;
      }
    }
}

// ---------------------------------------------------------------------------
// Kernel 5: dots2 = (q @ k^T) * SCALE per (b,h). M=N=1024, K=32 (one step).
// ---------------------------------------------------------------------------
__global__ __launch_bounds__(256)
void k_dots2(const bf16* __restrict__ pb, float* __restrict__ logits)
{
  const int bh = blockIdx.z, b = bh >> 3, h = bh & 7;
  const int m0 = blockIdx.x * BM, n0 = blockIdx.y * BN;
  const int tid = threadIdx.x, lane = tid & 31, wid = tid >> 5;
  const int wm = (wid >> 1) * 32, wn = (wid & 1) * 32;
  const bf16* q  = pb + ((size_t)b * 1024 + h * 32) * 1024;       // [c][token]
  const bf16* kk = pb + ((size_t)b * 1024 + 256 + h * 32) * 1024;
  __shared__ __align__(16) bf16 As[BM][LDP1];
  __shared__ __align__(16) bf16 Bt[BN][LDP1];
  v8f acc[2][2] = {};

  for (int u = tid; u < BM * 32; u += 256) {
    int m = u & 127, k = u >> 7;
    As[m][k] = q[(size_t)k * 1024 + m0 + m];
  }
  for (int u = tid; u < BN * 32; u += 256) {
    int n = u & 63, k = u >> 6;
    Bt[n][k] = kk[(size_t)k * 1024 + n0 + n];
  }
  __syncthreads();
  v16bf a0 = lds_frag_a<LDP1>(As, wm,      0, lane);
  v16bf a1 = lds_frag_a<LDP1>(As, wm + 16, 0, lane);
  v16bf b0 = lds_frag_b<LDP1>(Bt, wn,      0, lane);
  v16bf b1 = lds_frag_b<LDP1>(Bt, wn + 16, 0, lane);
  acc[0][0] = wmma_bf16(a0, b0, acc[0][0]);
  acc[0][1] = wmma_bf16(a0, b1, acc[0][1]);
  acc[1][0] = wmma_bf16(a1, b0, acc[1][0]);
  acc[1][1] = wmma_bf16(a1, b1, acc[1][1]);

  const int col = lane & 15, rb = (lane & 16) ? 8 : 0;
  #pragma unroll
  for (int i = 0; i < 2; i++)
    #pragma unroll
    for (int j = 0; j < 2; j++) {
      int iRow = m0 + wm + i * 16 + rb;
      int jCol = n0 + wn + j * 16 + col;
      #pragma unroll
      for (int r = 0; r < 8; r++)
        logits[((size_t)bh * 1024 + iRow + r) * 1024 + jCol] =
            acc[i][j][r] * 0.17677669529663687f;  // SCALE = dk^-0.5
    }
}

// ---------------------------------------------------------------------------
// Kernel 6: out = BN(Wo @ gelu_buf + bo; go, bo2). M=256, N=1024, K=1024.
// ---------------------------------------------------------------------------
__global__ __launch_bounds__(256)
void k_out(const float* __restrict__ Wo, const float* __restrict__ bo,
           const float* __restrict__ go, const float* __restrict__ bo2,
           const bf16* __restrict__ gbuf, float* __restrict__ out)
{
  const int b = blockIdx.z;
  const int m0 = blockIdx.x * BM, n0 = blockIdx.y * BN;
  const int tid = threadIdx.x, lane = tid & 31, wid = tid >> 5;
  const int wm = (wid >> 1) * 32, wn = (wid & 1) * 32;
  __shared__ __align__(16) bf16 As[BM][LDP2];
  __shared__ __align__(16) bf16 Bt[BN][LDP2];
  v8f acc[2][2] = {};

  for (int k0 = 0; k0 < 1024; k0 += 64) {
    for (int u = tid; u < BM * 64; u += 256) {
      int m = u >> 6, k = u & 63;
      As[m][k] = (bf16)Wo[(size_t)(m0 + m) * 1024 + k0 + k];
    }
    for (int u = tid; u < BN * 64; u += 256) {
      int n = u & 63, k = u >> 6;
      Bt[n][k] = gbuf[((size_t)b * 1024 + (k0 + k)) * 1024 + n0 + n];
    }
    if (k0 + 64 < 1024)
      __builtin_prefetch(&gbuf[((size_t)b * 1024 + (k0 + 64)) * 1024 + n0 + (tid & 63)], 0, 0);
    __syncthreads();
    WMMA_STEP(As, Bt, LDP2)
    __syncthreads();
  }

  const int col = lane & 15, rb = (lane & 16) ? 8 : 0;
  #pragma unroll
  for (int i = 0; i < 2; i++)
    #pragma unroll
    for (int j = 0; j < 2; j++) {
      int mB = m0 + wm + i * 16 + rb;
      int n  = n0 + wn + j * 16 + col;
      #pragma unroll
      for (int r = 0; r < 8; r++) {
        int m = mB + r;
        out[((size_t)b * 256 + m) * 1024 + n] = (acc[i][j][r] + bo[m]) * go[m] + bo2[m];
      }
    }
}

// ---------------------------------------------------------------------------
extern "C" void kernel_launch(void* const* d_in, const int* in_sizes, int n_in,
                              void* d_out, int out_size, void* d_ws, size_t ws_size,
                              hipStream_t stream) {
  (void)in_sizes; (void)n_in; (void)out_size; (void)ws_size;
  const float* x       = (const float*)d_in[0];
  const float* Wq      = (const float*)d_in[1];
  const float* gq      = (const float*)d_in[2];
  const float* bq      = (const float*)d_in[3];
  const float* Wk      = (const float*)d_in[4];
  const float* gk      = (const float*)d_in[5];
  const float* bk      = (const float*)d_in[6];
  const float* Wv      = (const float*)d_in[7];
  const float* gv      = (const float*)d_in[8];
  const float* bv      = (const float*)d_in[9];
  const float* Wmk     = (const float*)d_in[10];
  const float* gmk     = (const float*)d_in[11];
  const float* bmk     = (const float*)d_in[12];
  const float* pos_emb = (const float*)d_in[13];
  const float* Wo      = (const float*)d_in[14];
  const float* bo      = (const float*)d_in[15];
  const float* go      = (const float*)d_in[16];
  const float* bo2     = (const float*)d_in[17];

  // workspace layout (bytes):
  //   [0,   16M)  proj f32  (4 x 1024 x 1024)
  //   [16M, 24M)  proj bf16
  //   [24M, 32M)  gelu-out bf16 (4 x 1024ch x 1024tok)
  //   [32M, 160M) logits f32 (4 x 8 x 1024 x 1024), reused for both branches
  char* ws = (char*)d_ws;
  float* proj_f32 = (float*)(ws);
  bf16*  proj_b16 = (bf16*)(ws + ((size_t)16 << 20));
  bf16*  gbuf     = (bf16*)(ws + ((size_t)24 << 20));
  float* logits   = (float*)(ws + ((size_t)32 << 20));
  float* out      = (float*)d_out;

  const int avShmem = 2 * (BM + BN) * LDP2 * (int)sizeof(bf16);  // 55296 B (2 buffers)

  dim3 blk(256);
  k_proj  <<<dim3(8, 16, 4),  blk, 0, stream>>>(x, Wq, gq, bq, Wk, gk, bk, Wv, gv, bv,
                                                proj_f32, proj_b16);
  k_dots1 <<<dim3(64, 16, 4), blk, 0, stream>>>(Wmk, gmk, bmk, pos_emb, proj_f32, logits);
  k_softmax<<<dim3(32768),    blk, 0, stream>>>(logits);
  k_av    <<<dim3(8, 1, 32),  blk, avShmem, stream>>>(logits, proj_b16, gbuf, 0);
  k_dots2 <<<dim3(8, 16, 32), blk, 0, stream>>>(proj_b16, logits);
  k_softmax<<<dim3(32768),    blk, 0, stream>>>(logits);
  k_av    <<<dim3(8, 1, 32),  blk, avShmem, stream>>>(logits, proj_b16, gbuf, 512);
  k_out   <<<dim3(2, 16, 4),  blk, 0, stream>>>(Wo, bo, go, bo2, gbuf, out);
}
